// LegislativeGraphEncoder_34986803593676
// MI455X (gfx1250) — matrix operand order repaired and checked by reading, not compiled, not run on hardware
//
#include <hip/hip_runtime.h>
#include <hip/hip_bf16.h>

// ---------------------------------------------------------------------------
// CDNA5 (gfx1250) LegislativeGraphEncoder.
// GEMMs: v_wmma_f32_16x16x32_bf16 (wave32). Weights pre-packed fragment-major
// (one contiguous 32B load per B fragment). Wave pairs split a 32-row x 192-col
// band: each wave does 32 rows x 96 cols (96 accumulator VGPRs), so every B
// fragment feeds two WMMAs. Vote MLP fused via LDS staging. __launch_bounds__
// relaxes the occupancy cap so the register allocator does not spill.
// ---------------------------------------------------------------------------

typedef __attribute__((ext_vector_type(16))) __bf16 v16bf;
typedef __attribute__((ext_vector_type(8)))  __bf16 v8bf;
typedef __attribute__((ext_vector_type(8)))  float  v8f;

#define H 192
#define NT 12              // 192/16 N-tiles
#define N_DONOR 100000
#define N_LEG   5000
#define N_LOBBY 20000
#define N_BV    200000
#define N_BILL  100000
#define E_DON   1000000
#define E_LOB   500000
#define E_VER   200000
#define E_VOTE  300000

// ---- WMMA fragment helpers (ISA 7.12.2 layouts) ---------------------------
// Packed B: fragment f (= kc*NT + t), lane L holds 16 contiguous bf16.
__device__ __forceinline__ v16bf load_bp(const __bf16* Bp, int frag, int lane) {
  return *(const v16bf*)(Bp + (((size_t)frag * 32 + lane) << 4));
}
__device__ __forceinline__ v16bf pack16(float4 a0, float4 a1, float4 b0, float4 b1, float s) {
  v16bf a;
  a[0]=(__bf16)(a0.x*s); a[1]=(__bf16)(a0.y*s); a[2]=(__bf16)(a0.z*s); a[3]=(__bf16)(a0.w*s);
  a[4]=(__bf16)(a1.x*s); a[5]=(__bf16)(a1.y*s); a[6]=(__bf16)(a1.z*s); a[7]=(__bf16)(a1.w*s);
  a[8]=(__bf16)(b0.x*s); a[9]=(__bf16)(b0.y*s); a[10]=(__bf16)(b0.z*s); a[11]=(__bf16)(b0.w*s);
  a[12]=(__bf16)(b1.x*s); a[13]=(__bf16)(b1.y*s); a[14]=(__bf16)(b1.z*s); a[15]=(__bf16)(b1.w*s);
  return a;
}
// A fragment from a 32B-aligned f32 row (8+8 contiguous run at kb, kb+16)
__device__ __forceinline__ v16bf a_f32(const float* ap, int kb, float s) {
  float4 p0 = *(const float4*)(ap + kb);
  float4 p1 = *(const float4*)(ap + kb + 4);
  float4 p2 = *(const float4*)(ap + kb + 16);
  float4 p3 = *(const float4*)(ap + kb + 20);
  return pack16(p0, p1, p2, p3, s);
}
// A fragment from a 16B-aligned bf16 row (b128 pairs; works for LDS rows too)
__device__ __forceinline__ v16bf a_bf16(const __bf16* p, int kb) {
  v8bf lo = *(const v8bf*)(p + kb);
  v8bf hi = *(const v8bf*)(p + kb + 16);
  v16bf a;
#pragma unroll
  for (int e = 0; e < 8; ++e) { a[e] = lo[e]; a[e + 8] = hi[e]; }
  return a;
}

#define WMMA_BF16(a, b, c) \
  __builtin_amdgcn_wmma_f32_16x16x32_bf16(false, (a), false, (b), (short)0, (c), false, false)

// ---------------------------------------------------------------------------
__global__ void k_zero(float* p, size_t n) {
  size_t i = (size_t)blockIdx.x * blockDim.x + threadIdx.x;
  size_t stride = (size_t)gridDim.x * blockDim.x;
  for (; i < n; i += stride) p[i] = 0.f;
}

// Pack weight (src: 192 x K row-major) into fragment-major WMMA-B order.
__global__ void k_pack_b(const float* __restrict__ src, __bf16* __restrict__ dst,
                         int K, int Kchunks) {
  int idx = blockIdx.x * blockDim.x + threadIdx.x;
  int total = Kchunks * NT * 512;
  if (idx >= total) return;
  int e = idx & 15;
  int lane = (idx >> 4) & 31;
  int t = (idx >> 9) % NT;
  int kc = idx / (NT * 512);
  int k = kc * 32 + ((lane & 16) ? 8 : 0) + (e < 8 ? e : e + 8);
  int n = t * 16 + (lane & 15);
  dst[idx] = (k < K) ? (__bf16)src[(size_t)n * K + k] : (__bf16)0.f;
}

// per-edge temporal hidden: h_j = relu(ts*w1_j + b1_j), scatter-sum into acc[src]
__global__ void k_temporal_edges(const float* __restrict__ ts, const int* __restrict__ src,
                                 int E, const float* __restrict__ w1, const float* __restrict__ b1,
                                 float* acc, float* cnt) {
  int e = blockIdx.x * blockDim.x + threadIdx.x;
  if (e >= E) return;
  float t = ts[e];
  int s = src[e];
  float* dp = acc + (size_t)s * 48;
#pragma unroll 8
  for (int j = 0; j < 48; ++j) {
    float h = fmaxf(t * w1[j] + b1[j], 0.f);
    atomicAdd(dp + j, h);
  }
  atomicAdd(cnt + s, 1.f);
}

// xs = x + (acc/max(cnt,1)) @ w2^T + [cnt>0]*b2   (K=48 pad 64)
// Wave = 32-row band x 96-col half (nh selects N-tiles 0..5 or 6..11).
__global__ void __launch_bounds__(128, 1)
k_finish_xs(const float* __restrict__ x, const float* __restrict__ acc,
            const float* __restrict__ cnt, const __bf16* __restrict__ w2p,
            const float* __restrict__ b2, float* xs, int N) {
  int lane = threadIdx.x & 31;
  int wid = blockIdx.x * (blockDim.x >> 5) + (threadIdx.x >> 5);
  int band = wid >> 1, nh = wid & 1;
  int row0 = band * 32;
  if (row0 >= N) return;                     // wave-uniform exit (N % 32 == 0)
  int arow0 = row0 + (lane & 15);
  int arow1 = arow0 + 16;
  float inv0 = 1.f / fmaxf(cnt[arow0], 1.f);
  float inv1 = 1.f / fmaxf(cnt[arow1], 1.f);
  const float* ap0 = acc + (size_t)arow0 * 48;
  const float* ap1 = acc + (size_t)arow1 * 48;
  int kb = (lane & 16) ? 8 : 0;
  v8f d0[6], d1[6];
#pragma unroll
  for (int t = 0; t < 6; ++t)
#pragma unroll
    for (int i = 0; i < 8; ++i) { d0[t][i] = 0.f; d1[t][i] = 0.f; }
  {  // chunk 0: K 0..31
    v16bf a0 = a_f32(ap0, kb, inv0);
    v16bf a1 = a_f32(ap1, kb, inv1);
#pragma unroll
    for (int t = 0; t < 6; ++t) {
      v16bf b = load_bp(w2p, nh * 6 + t, lane);
      d0[t] = WMMA_BF16(a0, b, d0[t]);
      d1[t] = WMMA_BF16(a1, b, d1[t]);
    }
  }
  {  // chunk 1: K 32..47 valid, 48..63 zero
    float4 z = make_float4(0.f, 0.f, 0.f, 0.f);
    v16bf a0 = pack16(*(const float4*)(ap0 + 32 + kb), *(const float4*)(ap0 + 36 + kb), z, z, inv0);
    v16bf a1 = pack16(*(const float4*)(ap1 + 32 + kb), *(const float4*)(ap1 + 36 + kb), z, z, inv1);
#pragma unroll
    for (int t = 0; t < 6; ++t) {
      v16bf b = load_bp(w2p, NT + nh * 6 + t, lane);
      d0[t] = WMMA_BF16(a0, b, d0[t]);
      d1[t] = WMMA_BF16(a1, b, d1[t]);
    }
  }
  int col0 = lane & 15;
  int rloc = (lane & 16) ? 8 : 0;
#pragma unroll
  for (int half = 0; half < 2; ++half) {
    int rbase = row0 + half * 16 + rloc;
#pragma unroll
    for (int r = 0; r < 8; ++r) {
      int row = rbase + r;
      float sb = (cnt[row] > 0.f) ? 1.f : 0.f;
#pragma unroll
      for (int t = 0; t < 6; ++t) {
        int col = (nh * 6 + t) * 16 + col0;
        float dv = half ? d1[t][r] : d0[t][r];
        xs[(size_t)row * H + col] = x[(size_t)row * H + col] + dv + sb * b2[col];
      }
    }
  }
}

// gather xs[src], scatter-add into agg[dst] (+count). 32 lanes per edge.
__global__ void k_sage_edges(const float* __restrict__ xs, const int* __restrict__ src,
                             const int* __restrict__ dst, int E, float* agg, float* cnt) {
  size_t tid = (size_t)blockIdx.x * blockDim.x + threadIdx.x;
  int e = (int)(tid >> 5);
  int lane = (int)(tid & 31);
  if (e >= E) return;
  int s = src[e], d = dst[e];
  const float* xp = xs + (size_t)s * H;
  float* ap = agg + (size_t)d * H;
#pragma unroll
  for (int j = lane; j < H; j += 32) atomicAdd(ap + j, xp[j]);
  if (lane == 0) atomicAdd(cnt + d, 1.f);
}

// Fused vote MLP. One 64-thread block = one 32-row band; the two waves split N
// (96 cols each). Raw A and hidden h staged in LDS (bf16) with block barriers.
__global__ void __launch_bounds__(64, 1)
k_vote_fused(const float* __restrict__ va, const __bf16* __restrict__ w1p,
             const float* __restrict__ b1, const __bf16* __restrict__ w2p,
             const float* __restrict__ b2, const int* __restrict__ dstv,
             float* e_acc, float* e_cnt, int M) {
  __shared__ __align__(16) __bf16 sa[32 * 384];  // 24 KB raw features
  __shared__ __align__(16) __bf16 shh[32 * H];   // 12 KB hidden
  int lane = threadIdx.x & 31;
  int nh = threadIdx.x >> 5;                 // which 96-col half
  int row0 = blockIdx.x * 32;                // grid is exact (M % 32 == 0)
  int col0 = lane & 15;
  int rloc = (lane & 16) ? 8 : 0;
  int kb8 = (lane & 16) ? 8 : 0;
  // ---- stage raw A tile (coalesced f32 -> bf16) ----
  for (int idx = threadIdx.x; idx < 32 * 384; idx += 64) {
    int rr = idx / 384, cc = idx - rr * 384;
    sa[idx] = (__bf16)va[(size_t)(row0 + rr) * 385 + 1 + cc];
  }
  __syncthreads();
  v8f d0[6], d1[6];
#pragma unroll
  for (int t = 0; t < 6; ++t)
#pragma unroll
    for (int i = 0; i < 8; ++i) { d0[t][i] = 0.f; d1[t][i] = 0.f; }
  // ---- layer 1: raw(384) @ w1^T ----
  {
    const __bf16* a0p = sa + (lane & 15) * 384;
    const __bf16* a1p = sa + (16 + (lane & 15)) * 384;
    for (int kc = 0; kc < 12; ++kc) {
      int kb = kc * 32 + kb8;
      v16bf a0 = a_bf16(a0p, kb);
      v16bf a1 = a_bf16(a1p, kb);
#pragma unroll
      for (int t = 0; t < 6; ++t) {
        v16bf b = load_bp(w1p, kc * NT + nh * 6 + t, lane);
        d0[t] = WMMA_BF16(a0, b, d0[t]);
        d1[t] = WMMA_BF16(a1, b, d1[t]);
      }
    }
  }
  // ---- relu + bias, stage h tiles in LDS ----
#pragma unroll
  for (int r = 0; r < 8; ++r)
#pragma unroll
    for (int t = 0; t < 6; ++t) {
      int col = (nh * 6 + t) * 16 + col0;
      float bb = b1[col];
      shh[(rloc + r) * H + col]      = (__bf16)fmaxf(d0[t][r] + bb, 0.f);
      shh[(16 + rloc + r) * H + col] = (__bf16)fmaxf(d1[t][r] + bb, 0.f);
    }
  __syncthreads();
  // ---- layer 2: h(192) @ w2^T ----
#pragma unroll
  for (int t = 0; t < 6; ++t)
#pragma unroll
    for (int i = 0; i < 8; ++i) { d0[t][i] = 0.f; d1[t][i] = 0.f; }
  {
    const __bf16* hp0 = shh + (lane & 15) * H;
    const __bf16* hp1 = shh + (16 + (lane & 15)) * H;
#pragma unroll
    for (int kc = 0; kc < 6; ++kc) {
      int kb = kc * 32 + kb8;
      v16bf a0 = a_bf16(hp0, kb);
      v16bf a1 = a_bf16(hp1, kb);
#pragma unroll
      for (int t = 0; t < 6; ++t) {
        v16bf b = load_bp(w2p, kc * NT + nh * 6 + t, lane);
        d0[t] = WMMA_BF16(a0, b, d0[t]);
        d1[t] = WMMA_BF16(a1, b, d1[t]);
      }
    }
  }
  // ---- scale + scatter-mean accumulate ----
  if (nh == 0 && (lane & 15) == 0) {
#pragma unroll
    for (int r = 0; r < 8; ++r) {
      atomicAdd(&e_cnt[dstv[row0 + rloc + r]], 1.f);
      atomicAdd(&e_cnt[dstv[row0 + 16 + rloc + r]], 1.f);
    }
  }
#pragma unroll
  for (int half = 0; half < 2; ++half) {
    int rbase = row0 + half * 16 + rloc;
#pragma unroll
    for (int r = 0; r < 8; ++r) {
      int row = rbase + r;
      int dd = dstv[row];
      float pol = fminf(fmaxf(va[(size_t)row * 385], 0.f), 1.f);
      float s = pol + 0.01f;
#pragma unroll
      for (int t = 0; t < 6; ++t) {
        int col = (nh * 6 + t) * 16 + col0;
        float dv = half ? d1[t][r] : d0[t][r];
        atomicAdd(&e_acc[(size_t)dd * H + col], (dv + b2[col]) * s);
      }
    }
  }
}

// out = normalize( (agg/max(cnt,1)) @ wl^T + bl + xdst @ wr^T ) [+ residual]
// One 64-thread block = one 32-row band; waves split N (96 cols each).
// Row L2-norm partials exchanged through LDS.
__global__ void __launch_bounds__(64, 1)
k_sage_out(const float* __restrict__ agg, const float* __restrict__ cnt,
           const float* __restrict__ xdst, const __bf16* __restrict__ Bsp,
           const float* __restrict__ bl, float* out,
           const float* __restrict__ residual, int N) {
  __shared__ float snorm[2][32];
  int lane = threadIdx.x & 31;
  int nh = threadIdx.x >> 5;
  int row0 = blockIdx.x * 32;                // grid = ceil(N/32); whole block valid or tail
  int arow0 = row0 + (lane & 15);      if (arow0 > N - 1) arow0 = N - 1;
  int arow1 = row0 + 16 + (lane & 15); if (arow1 > N - 1) arow1 = N - 1;
  float inv0 = 1.f / fmaxf(cnt[arow0], 1.f);
  float inv1 = 1.f / fmaxf(cnt[arow1], 1.f);
  const float* ag0 = agg + (size_t)arow0 * H;
  const float* ag1 = agg + (size_t)arow1 * H;
  const float* xp0 = xdst + (size_t)arow0 * H;
  const float* xp1 = xdst + (size_t)arow1 * H;
  int half8 = (lane & 16) ? 8 : 0;
  v8f d0[6], d1[6];
#pragma unroll
  for (int t = 0; t < 6; ++t)
#pragma unroll
    for (int i = 0; i < 8; ++i) { d0[t][i] = 0.f; d1[t][i] = 0.f; }
#pragma unroll
  for (int kc = 0; kc < 6; ++kc) {           // K 0..191: mean-aggregated neighbors
    int kb = kc * 32 + half8;
    v16bf a0 = a_f32(ag0, kb, inv0);
    v16bf a1 = a_f32(ag1, kb, inv1);
#pragma unroll
    for (int t = 0; t < 6; ++t) {
      v16bf b = load_bp(Bsp, kc * NT + nh * 6 + t, lane);
      d0[t] = WMMA_BF16(a0, b, d0[t]);
      d1[t] = WMMA_BF16(a1, b, d1[t]);
    }
  }
#pragma unroll
  for (int kc = 6; kc < 12; ++kc) {          // K 192..383: self features
    int kb = (kc - 6) * 32 + half8;
    v16bf a0 = a_f32(xp0, kb, 1.f);
    v16bf a1 = a_f32(xp1, kb, 1.f);
#pragma unroll
    for (int t = 0; t < 6; ++t) {
      v16bf b = load_bp(Bsp, kc * NT + nh * 6 + t, lane);
      d0[t] = WMMA_BF16(a0, b, d0[t]);
      d1[t] = WMMA_BF16(a1, b, d1[t]);
    }
  }
  int col0 = lane & 15;
  int rloc = (lane & 16) ? 8 : 0;
  float p0[8], p1[8];
#pragma unroll
  for (int r = 0; r < 8; ++r) {
    float s0 = 0.f, s1 = 0.f;
#pragma unroll
    for (int t = 0; t < 6; ++t) {
      float bb = bl[(nh * 6 + t) * 16 + col0];
      d0[t][r] += bb; s0 += d0[t][r] * d0[t][r];
      d1[t][r] += bb; s1 += d1[t][r] * d1[t][r];
    }
    p0[r] = s0; p1[r] = s1;
  }
#pragma unroll
  for (int m = 1; m < 16; m <<= 1) {
#pragma unroll
    for (int r = 0; r < 8; ++r) {
      p0[r] += __shfl_xor(p0[r], m, 32);
      p1[r] += __shfl_xor(p1[r], m, 32);
    }
  }
  if ((lane & 15) == 0) {
#pragma unroll
    for (int r = 0; r < 8; ++r) {
      snorm[nh][rloc + r] = p0[r];
      snorm[nh][16 + rloc + r] = p1[r];
    }
  }
  __syncthreads();
#pragma unroll
  for (int half = 0; half < 2; ++half) {
    int rbase = row0 + half * 16 + rloc;
#pragma unroll
    for (int r = 0; r < 8; ++r) {
      int row = rbase + r;
      if (row < N) {
        float mine = half ? p1[r] : p0[r];
        float nr = mine + snorm[1 - nh][half * 16 + rloc + r];
        float is = 1.f / fmaxf(sqrtf(nr), 1e-12f);
#pragma unroll
        for (int t = 0; t < 6; ++t) {
          int col = (nh * 6 + t) * 16 + col0;
          float v = (half ? d1[t][r] : d0[t][r]) * is;
          if (residual) v += residual[(size_t)row * H + col];
          out[(size_t)row * H + col] = v;
        }
      }
    }
  }
}

__global__ void k_bv_out(const float* __restrict__ e_acc, const float* __restrict__ e_cnt,
                         const float* __restrict__ x_bv, float* out) {
  size_t i = (size_t)blockIdx.x * blockDim.x + threadIdx.x;
  if (i >= (size_t)N_BV * H) return;
  size_t row = i / H;
  out[i] = e_acc[i] / fmaxf(e_cnt[row], 1.f) + x_bv[i];
}

__global__ void k_combine_leg(const float* __restrict__ a, const float* __restrict__ b,
                              const float* __restrict__ x, float* out) {
  int i = blockIdx.x * blockDim.x + threadIdx.x;
  if (i >= N_LEG * H) return;
  out[i] = a[i] + b[i] + x[i];
}

// ---------------------------------------------------------------------------
extern "C" void kernel_launch(void* const* d_in, const int* in_sizes, int n_in,
                              void* d_out, int out_size, void* d_ws, size_t ws_size,
                              hipStream_t stream) {
  (void)in_sizes; (void)n_in; (void)out_size; (void)ws_size;
  // params flattened in dict insertion order
  const float* don_wl = (const float*)d_in[0];
  const float* don_bl = (const float*)d_in[1];
  const float* don_wr = (const float*)d_in[2];
  const float* lob_wl = (const float*)d_in[3];
  const float* lob_bl = (const float*)d_in[4];
  const float* lob_wr = (const float*)d_in[5];
  const float* ver_wl = (const float*)d_in[6];
  const float* ver_bl = (const float*)d_in[7];
  const float* ver_wr = (const float*)d_in[8];
  const float* tv_w1 = (const float*)d_in[9];   // t_vote
  const float* tv_b1 = (const float*)d_in[10];
  const float* tv_w2 = (const float*)d_in[11];
  const float* tv_b2 = (const float*)d_in[12];
  const float* td_w1 = (const float*)d_in[13];  // t_don
  const float* td_b1 = (const float*)d_in[14];
  const float* td_w2 = (const float*)d_in[15];
  const float* td_b2 = (const float*)d_in[16];
  const float* tl_w1 = (const float*)d_in[17];  // t_lob
  const float* tl_b1 = (const float*)d_in[18];
  const float* tl_w2 = (const float*)d_in[19];
  const float* tl_b2 = (const float*)d_in[20];
  const float* vm_w1 = (const float*)d_in[21];  // vote_mlp
  const float* vm_b1 = (const float*)d_in[22];
  const float* vm_w2 = (const float*)d_in[23];
  const float* vm_b2 = (const float*)d_in[24];
  const float* x_donor = (const float*)d_in[25];
  const float* x_leg   = (const float*)d_in[26];
  const float* x_lobby = (const float*)d_in[27];
  const float* x_bv    = (const float*)d_in[28];
  const float* x_bill  = (const float*)d_in[29];
  const float* ts_don = (const float*)d_in[30];
  const float* ts_lob = (const float*)d_in[31];
  const float* ts_ver = (const float*)d_in[32];
  const float* va     = (const float*)d_in[33];
  const int* src_don = (const int*)d_in[34];
  const int* dst_don = (const int*)d_in[35];
  const int* src_lob = (const int*)d_in[36];
  const int* dst_lob = (const int*)d_in[37];
  const int* src_ver = (const int*)d_in[38];
  const int* dst_ver = (const int*)d_in[39];
  const int* src_vote = (const int*)d_in[40];
  const int* dst_vote = (const int*)d_in[41];

  float* wsf = (float*)d_ws;
  // ---- workspace layout (float offsets; all multiples of 8 -> 32B aligned) ----
  constexpr size_t ACC_DON = 0;
  constexpr size_t CNT_DON = ACC_DON + (size_t)N_DONOR * 48;
  constexpr size_t ACC_LOB = CNT_DON + N_DONOR;
  constexpr size_t CNT_LOB = ACC_LOB + (size_t)N_LOBBY * 48;
  constexpr size_t ACC_VER = CNT_LOB + N_LOBBY;
  constexpr size_t CNT_VER = ACC_VER + (size_t)N_BV * 48;
  constexpr size_t AGG_DON = CNT_VER + N_BV;
  constexpr size_t CNT_ADON = AGG_DON + (size_t)N_LEG * H;
  constexpr size_t AGG_LOB = CNT_ADON + N_LEG;
  constexpr size_t CNT_ALOB = AGG_LOB + (size_t)N_LEG * H;
  constexpr size_t AGG_VER = CNT_ALOB + N_LEG;
  constexpr size_t CNT_AVER = AGG_VER + (size_t)N_BILL * H;
  constexpr size_t E_ACC = CNT_AVER + N_BILL;
  constexpr size_t E_CNT = E_ACC + (size_t)N_BV * H;
  constexpr size_t ZERO_TOTAL = E_CNT + N_BV;
  constexpr size_t XS_DON = ZERO_TOTAL;
  constexpr size_t XS_LOB = XS_DON + (size_t)N_DONOR * H;
  constexpr size_t XS_VER = XS_LOB + (size_t)N_LOBBY * H;
  constexpr size_t LEG_DON = XS_VER + (size_t)N_BV * H;
  constexpr size_t LEG_LOB = LEG_DON + (size_t)N_LEG * H;
  constexpr size_t BF_BASE = LEG_LOB + (size_t)N_LEG * H;   // bf16 region (32B aligned)

  __bf16* bfb = (__bf16*)(wsf + BF_BASE);
  __bf16* w1p_vm = bfb;                              // 12 chunks * 6144
  __bf16* w2p_vm = w1p_vm + 12 * NT * 512;           // 6 chunks
  __bf16* w2p_td = w2p_vm + 6 * NT * 512;            // 2 chunks (K=48 pad 64)
  __bf16* w2p_tl = w2p_td + 2 * NT * 512;
  __bf16* w2p_tv = w2p_tl + 2 * NT * 512;
  __bf16* bs_don = w2p_tv + 2 * NT * 512;            // 12 chunks (wl | wr)
  __bf16* bs_lob = bs_don + 12 * NT * 512;
  __bf16* bs_ver = bs_lob + 12 * NT * 512;

  float* out = (float*)d_out;
  constexpr size_t O2 = (size_t)N_DONOR * H;
  constexpr size_t O3 = O2 + (size_t)N_LEG * H;
  constexpr size_t O4 = O3 + (size_t)N_LOBBY * H;
  constexpr size_t O5 = O4 + (size_t)N_BV * H;

  // 1) zero accumulators
  k_zero<<<4096, 256, 0, stream>>>(wsf, ZERO_TOTAL);

  // 2) pack weights fragment-major bf16
  auto pk = [&](const float* s, __bf16* dsp, int K, int Kchunks) {
    int tot = Kchunks * NT * 512;
    k_pack_b<<<(tot + 255) / 256, 256, 0, stream>>>(s, dsp, K, Kchunks);
  };
  pk(vm_w1, w1p_vm, 384, 12);
  pk(vm_w2, w2p_vm, 192, 6);
  pk(td_w2, w2p_td, 48, 2);
  pk(tl_w2, w2p_tl, 48, 2);
  pk(tv_w2, w2p_tv, 48, 2);
  pk(don_wl, bs_don, 192, 6);  pk(don_wr, bs_don + 6 * NT * 512, 192, 6);
  pk(lob_wl, bs_lob, 192, 6);  pk(lob_wr, bs_lob + 6 * NT * 512, 192, 6);
  pk(ver_wl, bs_ver, 192, 6);  pk(ver_wr, bs_ver + 6 * NT * 512, 192, 6);

  // 3) temporal edge scatter (48-d hidden; GEMM hoisted to per-node)
  k_temporal_edges<<<(E_DON + 255) / 256, 256, 0, stream>>>(ts_don, src_don, E_DON, td_w1, td_b1, wsf + ACC_DON, wsf + CNT_DON);
  k_temporal_edges<<<(E_LOB + 255) / 256, 256, 0, stream>>>(ts_lob, src_lob, E_LOB, tl_w1, tl_b1, wsf + ACC_LOB, wsf + CNT_LOB);
  k_temporal_edges<<<(E_VER + 255) / 256, 256, 0, stream>>>(ts_ver, src_ver, E_VER, tv_w1, tv_b1, wsf + ACC_VER, wsf + CNT_VER);

  // 4) per-node temporal finish (WMMA; wave = 32 rows x 96 cols)
  auto fin = [&](const float* x, size_t acc, size_t cnt, __bf16* w2p, const float* b2, size_t xs, int N) {
    int waves = (N / 32) * 2;
    k_finish_xs<<<(waves + 3) / 4, 128, 0, stream>>>(x, wsf + acc, wsf + cnt, w2p, b2, wsf + xs, N);
  };
  fin(x_donor, ACC_DON, CNT_DON, w2p_td, td_b2, XS_DON, N_DONOR);
  fin(x_lobby, ACC_LOB, CNT_LOB, w2p_tl, tl_b2, XS_LOB, N_LOBBY);
  fin(x_bv,    ACC_VER, CNT_VER, w2p_tv, tv_b2, XS_VER, N_BV);

  // 5) SAGE edge aggregation (gather + L2-resident f32 atomics)
  k_sage_edges<<<(int)(((size_t)E_DON * 32 + 255) / 256), 256, 0, stream>>>(wsf + XS_DON, src_don, dst_don, E_DON, wsf + AGG_DON, wsf + CNT_ADON);
  k_sage_edges<<<(int)(((size_t)E_LOB * 32 + 255) / 256), 256, 0, stream>>>(wsf + XS_LOB, src_lob, dst_lob, E_LOB, wsf + AGG_LOB, wsf + CNT_ALOB);
  k_sage_edges<<<(int)(((size_t)E_VER * 32 + 255) / 256), 256, 0, stream>>>(wsf + XS_VER, src_ver, dst_ver, E_VER, wsf + AGG_VER, wsf + CNT_AVER);

  // 6) fused vote MLP: one 64-thread block per 32-row band (LDS-staged)
  k_vote_fused<<<E_VOTE / 32, 64, 0, stream>>>(va, w1p_vm, vm_b1, w2p_vm, vm_b2, dst_vote, wsf + E_ACC, wsf + E_CNT, E_VOTE);

  // 7) SAGE output GEMMs (fused K=384) + cross-wave L2 normalization
  k_sage_out<<<(N_LEG + 31) / 32, 64, 0, stream>>>(wsf + AGG_DON, wsf + CNT_ADON, x_leg, bs_don, don_bl, wsf + LEG_DON, nullptr, N_LEG);
  k_sage_out<<<(N_LEG + 31) / 32, 64, 0, stream>>>(wsf + AGG_LOB, wsf + CNT_ALOB, x_leg, bs_lob, lob_bl, wsf + LEG_LOB, nullptr, N_LEG);
  k_sage_out<<<(N_BILL + 31) / 32, 64, 0, stream>>>(wsf + AGG_VER, wsf + CNT_AVER, x_bill, bs_ver, ver_bl, out + O5, x_bill, N_BILL);

  // 8) epilogues
  k_bv_out<<<(int)(((size_t)N_BV * H + 255) / 256), 256, 0, stream>>>(wsf + E_ACC, wsf + E_CNT, x_bv, out + O4);
  k_combine_leg<<<(N_LEG * H + 255) / 256, 256, 0, stream>>>(wsf + LEG_DON, wsf + LEG_LOB, x_leg, out + O2);

  // 9) pass-through outputs
  hipMemcpyAsync(out, x_donor, (size_t)N_DONOR * H * sizeof(float), hipMemcpyDeviceToDevice, stream);
  hipMemcpyAsync(out + O3, x_lobby, (size_t)N_LOBBY * H * sizeof(float), hipMemcpyDeviceToDevice, stream);
}
